// LSTMEncoder2_46660524704388
// MI455X (gfx1250) — compile-verified, matrix operand order checked by reading, not run on hardware
//
#include <hip/hip_runtime.h>
#include <hip/hip_bf16.h>
#include <math.h>
#include <stdint.h>

// Problem constants (fixed by the reference).
#define BSZ 64
#define TSZ 512
#define DSZ 1024
#define HSZ 1024
#define G4  4096   // 4*H

typedef __bf16 bf16_t;
typedef __attribute__((ext_vector_type(16))) __bf16 v16bf;
typedef __attribute__((ext_vector_type(8)))  __bf16 v8bf;
typedef __attribute__((ext_vector_type(8)))  float  v8f;
typedef __attribute__((ext_vector_type(4)))  unsigned int v4u;
typedef __attribute__((ext_vector_type(8)))  int v8i;
typedef __attribute__((ext_vector_type(4)))  int v4i;

// Tensor Data Mover availability / arity detection.
#if defined(__has_builtin)
#  if __has_builtin(__builtin_amdgcn_tensor_load_to_lds)
#    define HAVE_TDM 1
#  endif
#endif
#if defined(HAVE_TDM)
#  if __has_include(<hip/amd_detail/amd_gfx1250_TDM.h>)
#    define TDM_SIX_ARGS 1
#  endif
#endif

// ---------------------------------------------------------------------------
// WMMA helpers
// ---------------------------------------------------------------------------

// Load a 16x32 bf16 fragment (A layout; B uses the mirrored layout with the
// operand stored transposed, so the same loader applies).
// Lane L (wave32): row/col = L&15 ; K chunks [k0..k0+7] and [k0+16..k0+23],
// k0 = (L>>4)*8.  `base` points at element (0, kk) of the tile, `ld` is the
// row stride in elements.  All loads are 16B aligned (ld multiple of 8).
__device__ __forceinline__ v16bf load_frag16x32(const bf16_t* __restrict__ base,
                                                int ld, int lane) {
  const int r  = lane & 15;
  const int k0 = (lane >> 4) << 3;
  const v8bf lo = *(const v8bf*)(base + (size_t)r * ld + k0);
  const v8bf hi = *(const v8bf*)(base + (size_t)r * ld + k0 + 16);
  return __builtin_shufflevector(lo, hi, 0, 1, 2, 3, 4, 5, 6, 7,
                                 8, 9, 10, 11, 12, 13, 14, 15);
}

__device__ __forceinline__ v8f wmma_bf16(v16bf a, v16bf b, v8f c) {
  // (neg_a, A, neg_b, B, c_mod, C, reuse_a, reuse_b)
  return __builtin_amdgcn_wmma_f32_16x16x32_bf16(false, a, false, b,
                                                 (short)0, c, false, false);
}

__device__ __forceinline__ float sigmoidf_(float x) {
  return 1.0f / (1.0f + __expf(-x));
}

// ---------------------------------------------------------------------------
// Tensor Data Mover: async copy of a 2D bf16 tile (tile_w x tile_h) from a
// row-major global tensor (row stride `row_stride` elements, logical size
// tensor_w x tensor_h for OOB) into contiguous LDS at byte offset lds_addr.
// One TDM op per call; tracked by TENSORcnt.
// ---------------------------------------------------------------------------
#if defined(HAVE_TDM)
__device__ __forceinline__ void tdm_load_tile(uint32_t lds_addr,
                                              const bf16_t* gsrc,
                                              int tile_w, int tile_h,
                                              int tensor_w, int tensor_h,
                                              int row_stride) {
  const uint64_t ga = (uint64_t)(uintptr_t)gsrc;
  v4u g0;
  g0.x = 1u;                                             // count=1, user mode
  g0.y = lds_addr;                                       // LDS byte address
  g0.z = (uint32_t)(ga & 0xffffffffu);                   // global_addr[31:0]
  g0.w = (uint32_t)((ga >> 32) & 0x01ffffffu) | (2u << 30);  // addr[56:32] | type=2
  v8i g1;
  g1[0] = (int)(1u << 16);                               // data_size=1 (2 bytes)
  g1[1] = (int)(((uint32_t)tensor_w & 0xffffu) << 16);   // tensor_dim0[15:0]
  g1[2] = (int)((((uint32_t)tensor_w >> 16) & 0xffffu) |
                (((uint32_t)tensor_h & 0xffffu) << 16)); // dim0[31:16] | dim1[15:0]
  g1[3] = (int)((((uint32_t)tensor_h >> 16) & 0xffffu) |
                (((uint32_t)tile_w & 0xffffu) << 16));   // dim1[31:16] | tile_dim0
  g1[4] = (int)((uint32_t)tile_h & 0xffffu);             // tile_dim1 | tile_dim2=0
  g1[5] = (int)(uint32_t)row_stride;                     // tensor_dim0_stride[31:0]
  g1[6] = 0;                                             // stride0[47:32] | stride1 lo
  g1[7] = 0;
  const v4i z4 = {0, 0, 0, 0};
#if defined(TDM_SIX_ARGS)
  const v8i z8 = {0, 0, 0, 0, 0, 0, 0, 0};
  __builtin_amdgcn_tensor_load_to_lds(g0, g1, z4, z4, z8, 0);
#else
  __builtin_amdgcn_tensor_load_to_lds(g0, g1, z4, z4, 0);
#endif
}
#endif  // HAVE_TDM

// ---------------------------------------------------------------------------
// Setup kernels
// ---------------------------------------------------------------------------

__global__ void zero_words_kernel(unsigned int* __restrict__ p, size_t nwords) {
  size_t i = (size_t)blockIdx.x * blockDim.x + threadIdx.x;
  if (i < nwords) p[i] = 0u;
}

// x (B,T,D) f32  ->  x_tm (T,B,D) bf16
__global__ void cast_x_tm_kernel(const float* __restrict__ x,
                                 bf16_t* __restrict__ x_tm) {
  size_t i = (size_t)blockIdx.x * blockDim.x + threadIdx.x;  // T*B*D outputs
  if (i >= (size_t)TSZ * BSZ * DSZ) return;
  int d = (int)(i % DSZ);
  size_t r = i / DSZ;
  int b = (int)(r % BSZ);
  int t = (int)(r / BSZ);
  x_tm[i] = (bf16_t)x[((size_t)b * TSZ + t) * DSZ + d];
}

// W (K,N) f32 -> Wt (N,K) bf16   (transpose + cast; one element per thread)
__global__ void transpose_cast_kernel(const float* __restrict__ W,
                                      bf16_t* __restrict__ Wt,
                                      int K, int N) {
  size_t i = (size_t)blockIdx.x * blockDim.x + threadIdx.x;  // N*K outputs
  if (i >= (size_t)N * K) return;
  int k = (int)(i % K);
  int n = (int)(i / K);
  Wt[i] = (bf16_t)W[(size_t)k * N + n];
}

// ---------------------------------------------------------------------------
// Batched input projection:  C[M,N] = A[M,K] @ Bt[N,K]^T + bias[N]
// Block = 256 threads = 8 waves; block tile = 64x64; wave = two 16x16 tiles
// sharing one A fragment.  bf16 WMMA, f32 accumulate.
// ---------------------------------------------------------------------------
__global__ __launch_bounds__(256) void gemm_bias_kernel(
    const bf16_t* __restrict__ A,   // M x K, row-major bf16
    const bf16_t* __restrict__ Bt,  // N x K, row-major bf16 (i.e. B^T)
    const float*  __restrict__ bias,
    float* __restrict__ C,          // M x N f32
    int M, int N, int K) {
  const int wave = threadIdx.x >> 5;
  const int lane = threadIdx.x & 31;
  const int mt  = wave >> 1;            // 0..3
  const int ntp = (wave & 1) << 1;      // 0 or 2
  const int m0 = blockIdx.y * 64 + mt * 16;
  const int n0 = blockIdx.x * 64 + ntp * 16;

  const bf16_t* Arow = A  + (size_t)m0 * K;
  const bf16_t* Bc0  = Bt + (size_t)n0 * K;
  const bf16_t* Bc1  = Bt + (size_t)(n0 + 16) * K;

  v8f acc0 = {};
  v8f acc1 = {};
  for (int kk = 0; kk < K; kk += 32) {
    v16bf af = load_frag16x32(Arow + kk, K, lane);
    v16bf b0 = load_frag16x32(Bc0 + kk, K, lane);
    v16bf b1 = load_frag16x32(Bc1 + kk, K, lane);
    acc0 = wmma_bf16(af, b0, acc0);
    acc1 = wmma_bf16(af, b1, acc1);
  }

  // D layout: lane (0..15) -> col = lane, rows r + 0 ; lane (16..31) -> rows r + 8.
  const int col   = n0 + (lane & 15);
  const int rbase = m0 + ((lane >> 4) << 3);
  const float bi0 = bias[col];
  const float bi1 = bias[col + 16];
#pragma unroll
  for (int r = 0; r < 8; ++r) {
    C[(size_t)(rbase + r) * N + col]      = acc0[r] + bi0;
    C[(size_t)(rbase + r) * N + col + 16] = acc1[r] + bi1;
  }
}

// ---------------------------------------------------------------------------
// One LSTM timestep (fused recurrent GEMM + gates), launched once per t.
//   z = xz_t + h_in @ U     (M=64, N=4096, K=1024, bf16 WMMA)
// Grid: 64 blocks; block `bid` owns h columns [bid*16, bid*16+16) for all 4
// gates (elementwise cell update stays block-local).  8 waves: wave w ->
// gate g = w>>1, M-tiles (w&1)*2 + {0,1}.
//
// h (64x1024 bf16, 128 KB) is staged into LDS in 4 chunks of 64x256 (32 KB),
// double-buffered: wave 0 issues a TDM tensor_load_to_lds for chunk i+1,
// waits TENSORcnt<=1 (chunk i complete, TDM ops retire in order), then a
// workgroup barrier releases the compute waves.  A fragments come from LDS
// (ds_load_b128); U fragments stream from L2.  Removes the 4x intra-block
// redundancy on h reads that bounds the latency-critical scan.
// ---------------------------------------------------------------------------
#define KCH 256                      // K chunk (elements)
#define NCH (HSZ / KCH)              // 4 chunks

__global__ __launch_bounds__(256) void lstm_step_kernel(
    const float*  __restrict__ xz_t,      // B x 4H slice for this t
    const bf16_t* __restrict__ Ut,        // 4H x H (U transposed), bf16
    const bf16_t* __restrict__ h_in,      // B x H bf16 (previous step)
    bf16_t*       __restrict__ h_out_bf,  // B x H bf16 (next step input)
    float*        __restrict__ h_state,   // B x H f32 (masked carry)
    float*        __restrict__ c_state,   // B x H f32
    const int*    __restrict__ lengths,   // B
    float*        __restrict__ out_f32,   // layer1: rnnout (B,T,H), else null
    bf16_t*       __restrict__ out_bf,    // layer0: out0_tm (T,B,H), else null
    int t) {
  __shared__ bf16_t hsh[2][BSZ][KCH];     // 2 x 32 KB ping-pong h staging
  __shared__ float zbuf[4][BSZ][16];      // 16 KB gate staging

  const int hc   = blockIdx.x * 16;       // h-column block
  const int wave = threadIdx.x >> 5;
  const int lane = threadIdx.x & 31;
  const int g    = wave >> 1;             // gate 0..3 (i,f,g,o)
  const int mt0  = (wave & 1) << 1;       // 0 or 2

  const bf16_t* Brow = Ut + (size_t)(g * HSZ + hc) * HSZ;  // 16 x K

#if defined(HAVE_TDM)
  const bool issuer = (wave == 0);
  if (issuer) {
    tdm_load_tile((uint32_t)(uintptr_t)&hsh[0][0][0], h_in,
                  KCH, BSZ, HSZ, BSZ, HSZ);
  }
#endif

  v8f acc0 = {};
  v8f acc1 = {};
  for (int ci = 0; ci < NCH; ++ci) {
    const int buf = ci & 1;
#if defined(HAVE_TDM)
    if (issuer) {
      if (ci + 1 < NCH) {
        tdm_load_tile((uint32_t)(uintptr_t)&hsh[buf ^ 1][0][0],
                      h_in + (ci + 1) * KCH, KCH, BSZ, HSZ, BSZ, HSZ);
        __builtin_amdgcn_s_wait_tensorcnt((short)1);   // chunk ci landed
      } else {
        __builtin_amdgcn_s_wait_tensorcnt((short)0);
      }
    }
    __syncthreads();                      // chunk ci visible to all waves
#else
    // Fallback: cooperative vector copy of chunk ci into hsh[buf].
    for (int idx = threadIdx.x; idx < BSZ * KCH / 8; idx += 256) {
      const int row = idx >> 5;           // KCH/8 = 32 v8bf per row
      const int c8  = idx & 31;
      *(v8bf*)&hsh[buf][row][c8 * 8] =
          *(const v8bf*)(h_in + (size_t)row * HSZ + ci * KCH + c8 * 8);
    }
    __syncthreads();
#endif

    const bf16_t* Ab0 = &hsh[buf][mt0 * 16][0];
    const bf16_t* Ab1 = &hsh[buf][mt0 * 16 + 16][0];
    const bf16_t* Bk  = Brow + ci * KCH;
#pragma unroll 4
    for (int kl = 0; kl < KCH; kl += 32) {
      v16bf bfr = load_frag16x32(Bk + kl, HSZ, lane);
      v16bf a0  = load_frag16x32(Ab0 + kl, KCH, lane);
      v16bf a1  = load_frag16x32(Ab1 + kl, KCH, lane);
      acc0 = wmma_bf16(a0, bfr, acc0);
      acc1 = wmma_bf16(a1, bfr, acc1);
    }
    __syncthreads();                      // done reading hsh[buf] before reuse
  }

  // z = acc + xz  -> LDS, grouped by gate.
  const int col  = lane & 15;
  const int rb   = (lane >> 4) << 3;
  const int zcol = g * HSZ + hc + col;
#pragma unroll
  for (int r = 0; r < 8; ++r) {
    const int m0 = mt0 * 16 + rb + r;
    const int m1 = m0 + 16;
    zbuf[g][m0][col] = acc0[r] + xz_t[(size_t)m0 * G4 + zcol];
    zbuf[g][m1][col] = acc1[r] + xz_t[(size_t)m1 * G4 + zcol];
  }
  __syncthreads();

  // Elementwise cell update: 256 threads x 4 elements = 64x16.
  const int j  = threadIdx.x & 15;
  const int mb = (threadIdx.x >> 4) << 2;
#pragma unroll
  for (int k = 0; k < 4; ++k) {
    const int m   = mb + k;
    const int idx = m * HSZ + hc + j;
    const float iv = sigmoidf_(zbuf[0][m][j]);
    const float fv = sigmoidf_(zbuf[1][m][j]);
    const float gv = tanhf(zbuf[2][m][j]);
    const float ov = sigmoidf_(zbuf[3][m][j]);
    const float c_old = c_state[idx];
    const float c_new = fv * c_old + iv * gv;
    const float h_new = ov * tanhf(c_new);
    const bool  msk   = t < lengths[m];
    const float h2 = msk ? h_new : h_state[idx];
    const float c2 = msk ? c_new : c_old;
    h_state[idx]  = h2;
    c_state[idx]  = c2;
    h_out_bf[idx] = (bf16_t)h2;
    if (out_bf)  out_bf[(size_t)t * BSZ * HSZ + idx] = (bf16_t)h2;       // (T,B,H)
    if (out_f32) out_f32[((size_t)m * TSZ + t) * HSZ + hc + j] = h2;     // (B,T,H)
  }
}

// ---------------------------------------------------------------------------
// Host-side launch
// ---------------------------------------------------------------------------
extern "C" void kernel_launch(void* const* d_in, const int* in_sizes, int n_in,
                              void* d_out, int out_size, void* d_ws, size_t ws_size,
                              hipStream_t stream) {
  (void)in_sizes; (void)n_in; (void)out_size; (void)ws_size;
  const float* x       = (const float*)d_in[0];
  const int*   lengths = (const int*)d_in[1];
  const float* W0 = (const float*)d_in[2];
  const float* U0 = (const float*)d_in[3];
  const float* b0 = (const float*)d_in[4];
  const float* W1 = (const float*)d_in[5];
  const float* U1 = (const float*)d_in[6];
  const float* b1 = (const float*)d_in[7];
  float* rnnout = (float*)d_out;                                   // (B,T,H)
  float* h_out  = rnnout + (size_t)BSZ * TSZ * HSZ;                // (B,H)
  float* c_out  = h_out + (size_t)BSZ * HSZ;                       // (B,H)

  // Workspace carve-up (all offsets 256B-aligned).
  char* ws = (char*)d_ws;
  size_t off = 0;
  auto carve = [&](size_t bytes) {
    void* p = ws + off;
    off += (bytes + 255) & ~(size_t)255;
    return p;
  };
  bf16_t* x_tm    = (bf16_t*)carve((size_t)TSZ * BSZ * DSZ * 2);   // 64 MB
  bf16_t* out0_tm = (bf16_t*)carve((size_t)TSZ * BSZ * HSZ * 2);   // 64 MB
  bf16_t* W0t     = (bf16_t*)carve((size_t)G4 * DSZ * 2);          // 8 MB
  bf16_t* U0t     = (bf16_t*)carve((size_t)G4 * HSZ * 2);          // 8 MB
  bf16_t* W1t     = (bf16_t*)carve((size_t)G4 * HSZ * 2);          // 8 MB
  bf16_t* U1t     = (bf16_t*)carve((size_t)G4 * HSZ * 2);          // 8 MB
  float*  xz      = (float*)carve((size_t)TSZ * BSZ * G4 * 4);     // 512 MB (reused per layer)
  float*  h_state = (float*)carve((size_t)BSZ * HSZ * 4);
  float*  c_state = (float*)carve((size_t)BSZ * HSZ * 4);
  bf16_t* hbuf    = (bf16_t*)carve((size_t)2 * BSZ * HSZ * 2);     // ping-pong

  const size_t BH = (size_t)BSZ * HSZ;

  // --- Setup: casts/transposes ---
  {
    size_t n = (size_t)TSZ * BSZ * DSZ;
    cast_x_tm_kernel<<<(unsigned)((n + 255) / 256), 256, 0, stream>>>(x, x_tm);
    size_t nw = (size_t)G4 * DSZ;
    transpose_cast_kernel<<<(unsigned)((nw + 255) / 256), 256, 0, stream>>>(W0, W0t, DSZ, G4);
    size_t nu = (size_t)G4 * HSZ;
    transpose_cast_kernel<<<(unsigned)((nu + 255) / 256), 256, 0, stream>>>(U0, U0t, HSZ, G4);
    transpose_cast_kernel<<<(unsigned)((nu + 255) / 256), 256, 0, stream>>>(W1, W1t, HSZ, G4);
    transpose_cast_kernel<<<(unsigned)((nu + 255) / 256), 256, 0, stream>>>(U1, U1t, HSZ, G4);
  }

  auto zero_states = [&]() {
    // h_state + c_state (f32) + hbuf (2x bf16) = contiguous word-zeroable region
    size_t nwords = (BH * 4 * 2 + 2 * BH * 2) / 4;
    zero_words_kernel<<<(unsigned)((nwords + 255) / 256), 256, 0, stream>>>(
        (unsigned int*)h_state, nwords);
  };

  const int M = TSZ * BSZ;             // 32768
  dim3 gemm_grid(G4 / 64, M / 64);     // (64, 512)

  // ===== Layer 0 =====
  zero_states();
  gemm_bias_kernel<<<gemm_grid, 256, 0, stream>>>(x_tm, W0t, b0, xz, M, G4, DSZ);
  for (int t = 0; t < TSZ; ++t) {
    const bf16_t* hin  = hbuf + (size_t)(t & 1) * BH;
    bf16_t*       hout = hbuf + (size_t)((t + 1) & 1) * BH;
    lstm_step_kernel<<<HSZ / 16, 256, 0, stream>>>(
        xz + (size_t)t * BSZ * G4, U0t, hin, hout, h_state, c_state,
        lengths, nullptr, out0_tm, t);
  }

  // ===== Layer 1 =====
  zero_states();
  gemm_bias_kernel<<<gemm_grid, 256, 0, stream>>>(out0_tm, W1t, b1, xz, M, G4, HSZ);
  for (int t = 0; t < TSZ; ++t) {
    const bf16_t* hin  = hbuf + (size_t)(t & 1) * BH;
    bf16_t*       hout = hbuf + (size_t)((t + 1) & 1) * BH;
    lstm_step_kernel<<<HSZ / 16, 256, 0, stream>>>(
        xz + (size_t)t * BSZ * G4, U1t, hin, hout, h_state, c_state,
        lengths, rnnout, nullptr, t);
  }

  // Final h, c of layer 1 -> output tail.
  hipMemcpyAsync(h_out, h_state, BH * sizeof(float), hipMemcpyDeviceToDevice, stream);
  hipMemcpyAsync(c_out, c_state, BH * sizeof(float), hipMemcpyDeviceToDevice, stream);
}
